// SampleNetFullResidual_87866440941657
// MI455X (gfx1250) — compile-verified, matrix operand order checked by reading, not run on hardware
//
#include <hip/hip_runtime.h>
#include <hip/hip_bf16.h>

// ---------------------------------------------------------------------------
// SampleNetFullResidual on MI455X (gfx1250, wave32, WMMA)
//
// Main kernel: one workgroup per pooling segment (1024 contiguous rows).
//   8 waves x 16 rows, tiles of 128 rows. Fused: nn (128->128->128->32),
//   gather(graph_out) add, gm1 (32-wide), in-register mean pool.
//   bf16 WMMA 16x16x32 with f32 accumulation, all weights in LDS (transposed
//   [N][K] so B fragments are contiguous ds_load_b32's). Next-tile rows are
//   prefetched (global_prefetch_b8) while the current tile computes.
// Tail kernel: tiny per-row fp32 chain (gm2/fm/out/out_lin) -> d_out (1024x2).
// ---------------------------------------------------------------------------

typedef __attribute__((ext_vector_type(16))) __bf16 v16bf;
typedef __attribute__((ext_vector_type(8)))  float  v8f;

#define LDA   132   // K-stride (elems) for 128-wide mats & activation tiles
#define LDW2  36    // K-stride (elems) for 32-wide mats

// --- LDS layout (bf16 elements) ---
enum : int {
  OFF_W1A  = 0,
  OFF_W2A  = OFF_W1A + 128 * LDA,
  OFF_W1B  = OFF_W2A + 128 * LDA,
  OFF_W2B  = OFF_W1B + 128 * LDA,
  OFF_W1C  = OFF_W2B + 128 * LDA,     // 16 rows x LDA   (128->16)
  OFF_W2C  = OFF_W1C + 16 * LDA,      // 32 rows x LDW2  (16->32, K padded)
  OFF_G0W1 = OFF_W2C + 32 * LDW2,
  OFF_G0W2 = OFF_G0W1 + 32 * LDW2,
  OFF_G1W1 = OFF_G0W2 + 32 * LDW2,
  OFF_G1W2 = OFF_G1W1 + 32 * LDW2,
  OFF_G2W1 = OFF_G1W2 + 32 * LDW2,    // 16 rows x LDW2  (32->16)
  OFF_G2W2 = OFF_G2W1 + 16 * LDW2,    // 32 rows x LDW2  (16->32, K padded)
  OFF_XT   = OFF_G2W2 + 32 * LDW2,    // 128 x LDA activation tile
  OFF_HT   = OFF_XT + 128 * LDA,      // 128 x LDA activation tile
  BF_ELEMS = OFF_HT + 128 * LDA
};
// --- f32 region (float elements), after the bf16 region ---
enum : int {
  FB_B1A = 0,   FB_B2A = 128, FB_B1B = 256, FB_B2B = 384,
  FB_B1C = 512, FB_B2C = 528,
  FB_G0B1 = 560, FB_G0B2 = 592, FB_G1B1 = 624, FB_G1B2 = 656,
  FB_G2B1 = 688, FB_G2B2 = 704,
  FB_RED  = 736,                       // 8 waves x 32 cols
  F32_ELEMS = FB_RED + 8 * 32
};
#define SMEM_BYTES (BF_ELEMS * 2 + F32_ELEMS * 4)   // ~225,920 B (<320KB/WGP)

__device__ __forceinline__ float elu_(float x) {
  return x > 0.f ? x : (__expf(x) - 1.f);
}

// Load a 16x32 bf16 fragment (A layout per ISA 7.12.2; B layout mirrored with
// N<->M since weights are stored transposed [N][K]). 8x ds_load_b32 per frag.
__device__ __forceinline__ v16bf frag(const __bf16* buf, int ld, int k0, int lane) {
  const __bf16* pp = buf + (lane & 15) * ld + k0 + ((lane >> 4) << 3);
  union { v16bf v; unsigned u[8]; } f;
#pragma unroll
  for (int i = 0; i < 8; ++i) {
    int k = ((i & 3) << 1) + ((i >> 2) << 4);   // pairs: 0,2,4,6,16,18,20,22
    f.u[i] = *(const unsigned*)(pp + k);
  }
  return f.v;
}

__device__ __forceinline__ v8f wmma_bf16(v16bf a, v16bf b, v8f c) {
  return __builtin_amdgcn_wmma_f32_16x16x32_bf16(
      false, a, false, b, (short)0, c, false, false);
}

// One dense layer over this wave's 16 rows:
//   O = elu(A @ W^T + bias [+ resid]), NT n-tiles of 16, KT k-steps of 32.
template <int NT, int KT>
__device__ __forceinline__ void mlp_layer(const __bf16* Ab, __bf16* Ob,
                                          const __bf16* Wt, int ldw,
                                          const float* bias,
                                          const __bf16* resid, int lane,
                                          bool zero_hi = false) {
  const int col = lane & 15, half = lane >> 4;
  v16bf a[KT];
#pragma unroll
  for (int k = 0; k < KT; ++k) a[k] = frag(Ab, LDA, k * 32, lane);
#pragma unroll
  for (int n = 0; n < NT; ++n) {
    float bv = bias[n * 16 + col];
    v8f c;
#pragma unroll
    for (int r = 0; r < 8; ++r) c[r] = bv;
#pragma unroll
    for (int k = 0; k < KT; ++k)
      c = wmma_bf16(a[k], frag(Wt + n * 16 * ldw, ldw, k * 32, lane), c);
#pragma unroll
    for (int r = 0; r < 8; ++r) {
      int m = r + half * 8;
      float v = c[r];
      if (resid) v += (float)resid[m * LDA + n * 16 + col];
      v = elu_(v);
      Ob[m * LDA + n * 16 + col] = (__bf16)v;
    }
  }
  if (zero_hi) {   // zero K-pad columns [NT*16, NT*16+16) for the next GEMM
#pragma unroll
    for (int r = 0; r < 8; ++r)
      Ob[(r + half * 8) * LDA + NT * 16 + col] = (__bf16)0.f;
  }
}

// Cooperative global f32 -> LDS bf16, transposed to [N][K] with K zero-pad.
__device__ __forceinline__ void loadW(__bf16* dst, const float* src,
                                      int K, int N, int Kpad, int ld, int tid) {
  int total = N * Kpad;
  for (int i = tid; i < total; i += 256) {
    int n = i / Kpad, k = i - n * Kpad;
    float v = (k < K) ? src[k * N + n] : 0.f;
    dst[n * ld + k] = (__bf16)v;
  }
}
__device__ __forceinline__ void loadB(float* dst, const float* src, int n, int tid) {
  for (int i = tid; i < n; i += 256) dst[i] = src[i];
}

struct MainArgs {
  const float* x;     // sample_node_feature (E x 128)
  const int*   nid;   // sample_node_id (E x 2): [sid, node_idx]
  const float* g;     // graph_out (20000 x 32)
  // w/b order: nn0.l1, nn0.l2, nn1.l1, nn1.l2, nn2.l1, nn2.l2,
  //            gm1_0.l1, gm1_0.l2, gm1_1.l1, gm1_1.l2, gm1_2.l1, gm1_2.l2
  const float* w[12];
  const float* b[12];
  float* pooled;      // workspace: (1024 x 32) f32
};

__global__ __launch_bounds__(256) void segment_mlp_kernel(MainArgs p) {
  extern __shared__ char smem[];
  __bf16* SB = (__bf16*)smem;
  float*  SF = (float*)(smem + BF_ELEMS * 2);

  const int tid  = threadIdx.x;
  const int lane = tid & 31;
  const int wave = tid >> 5;
  const int seg  = blockIdx.x;
  const int col  = lane & 15;
  const int half = lane >> 4;

  // ---- stage 0: weights -> LDS (once per workgroup; L2-resident source) ----
  loadW(SB + OFF_W1A,  p.w[0], 128, 128, 128, LDA,  tid);
  loadW(SB + OFF_W2A,  p.w[1], 128, 128, 128, LDA,  tid);
  loadW(SB + OFF_W1B,  p.w[2], 128, 128, 128, LDA,  tid);
  loadW(SB + OFF_W2B,  p.w[3], 128, 128, 128, LDA,  tid);
  loadW(SB + OFF_W1C,  p.w[4], 128, 16, 128, LDA,  tid);
  loadW(SB + OFF_W2C,  p.w[5],  16, 32,  32, LDW2, tid);
  loadW(SB + OFF_G0W1, p.w[6],  32, 32,  32, LDW2, tid);
  loadW(SB + OFF_G0W2, p.w[7],  32, 32,  32, LDW2, tid);
  loadW(SB + OFF_G1W1, p.w[8],  32, 32,  32, LDW2, tid);
  loadW(SB + OFF_G1W2, p.w[9],  32, 32,  32, LDW2, tid);
  loadW(SB + OFF_G2W1, p.w[10], 32, 16,  32, LDW2, tid);
  loadW(SB + OFF_G2W2, p.w[11], 16, 32,  32, LDW2, tid);
  loadB(SF + FB_B1A,  p.b[0], 128, tid);
  loadB(SF + FB_B2A,  p.b[1], 128, tid);
  loadB(SF + FB_B1B,  p.b[2], 128, tid);
  loadB(SF + FB_B2B,  p.b[3], 128, tid);
  loadB(SF + FB_B1C,  p.b[4],  16, tid);
  loadB(SF + FB_B2C,  p.b[5],  32, tid);
  loadB(SF + FB_G0B1, p.b[6],  32, tid);
  loadB(SF + FB_G0B2, p.b[7],  32, tid);
  loadB(SF + FB_G1B1, p.b[8],  32, tid);
  loadB(SF + FB_G1B2, p.b[9],  32, tid);
  loadB(SF + FB_G2B1, p.b[10], 16, tid);
  loadB(SF + FB_G2B2, p.b[11], 32, tid);
  __syncthreads();

  // Per-wave exclusive activation rows => no barriers in the main loop.
  __bf16* Xt = SB + OFF_XT + wave * 16 * LDA;
  __bf16* Ht = SB + OFF_HT + wave * 16 * LDA;

  float sum0 = 0.f, sum1 = 0.f;   // running column sums for mean pool

  for (int t = 0; t < 8; ++t) {
    const int gr0 = seg * 1024 + t * 128 + wave * 16;  // wave's first row

    // ---- load this wave's 16 rows of X (f32 -> bf16) ----
#pragma unroll
    for (int r = 0; r < 16; ++r) {
      const float4 v = ((const float4*)(p.x + (long)(gr0 + r) * 128))[lane];
      __bf16* d = Xt + r * LDA + lane * 4;
      d[0] = (__bf16)v.x; d[1] = (__bf16)v.y;
      d[2] = (__bf16)v.z; d[3] = (__bf16)v.w;
    }

    // ---- prefetch next tile's rows into cache while this tile computes ----
    if (t < 7) {
      const float* nx = p.x + (long)(gr0 + 128) * 128 + lane * 4;
#pragma unroll
      for (int r = 0; r < 16; ++r)
        __builtin_prefetch(nx + r * 128, 0, 1);   // -> global_prefetch_b8
    }

    // ---- nn block 0 (128->128->128, residual) ----
    mlp_layer<8, 4>(Xt, Ht, SB + OFF_W1A, LDA, SF + FB_B1A, nullptr, lane);
    mlp_layer<8, 4>(Ht, Xt, SB + OFF_W2A, LDA, SF + FB_B2A, Xt,      lane);
    // ---- nn block 1 ----
    mlp_layer<8, 4>(Xt, Ht, SB + OFF_W1B, LDA, SF + FB_B1B, nullptr, lane);
    mlp_layer<8, 4>(Ht, Xt, SB + OFF_W2B, LDA, SF + FB_B2B, Xt,      lane);
    // ---- nn block 2 l1: 128 -> 16 (zero pad cols 16..31 for next K=32) ----
    mlp_layer<1, 4>(Xt, Ht, SB + OFF_W1C, LDA, SF + FB_B1C, nullptr, lane, true);

    // ---- nn block 2 l2: 16(pad 32) -> 32, elu, then + graph gather -> Xt ----
    {
      v16bf a = frag(Ht, LDA, 0, lane);
      int nodes[8];
#pragma unroll
      for (int r = 0; r < 8; ++r)
        nodes[r] = p.nid[2 * (gr0 + r + half * 8) + 1];
#pragma unroll
      for (int nt = 0; nt < 2; ++nt) {
        float bv = SF[FB_B2C + nt * 16 + col];
        v8f c;
#pragma unroll
        for (int r = 0; r < 8; ++r) c[r] = bv;
        c = wmma_bf16(a, frag(SB + OFF_W2C + nt * 16 * LDW2, LDW2, 0, lane), c);
#pragma unroll
        for (int r = 0; r < 8; ++r) {
          float v = elu_(c[r]) + p.g[nodes[r] * 32 + nt * 16 + col];
          Xt[(r + half * 8) * LDA + nt * 16 + col] = (__bf16)v;
        }
      }
    }

    // ---- gm1 block 0 (32->32->32, residual) ----
    mlp_layer<2, 1>(Xt, Ht, SB + OFF_G0W1, LDW2, SF + FB_G0B1, nullptr, lane);
    mlp_layer<2, 1>(Ht, Xt, SB + OFF_G0W2, LDW2, SF + FB_G0B2, Xt,      lane);
    // ---- gm1 block 1 ----
    mlp_layer<2, 1>(Xt, Ht, SB + OFF_G1W1, LDW2, SF + FB_G1B1, nullptr, lane);
    mlp_layer<2, 1>(Ht, Xt, SB + OFF_G1W2, LDW2, SF + FB_G1B2, Xt,      lane);
    // ---- gm1 block 2 l1: 32 -> 16 (pad) ----
    mlp_layer<1, 1>(Xt, Ht, SB + OFF_G2W1, LDW2, SF + FB_G2B1, nullptr, lane, true);
    // ---- gm1 block 2 l2: 16(pad)->32 + residual, elu, accumulate pool ----
    {
      v16bf a = frag(Ht, LDA, 0, lane);
#pragma unroll
      for (int nt = 0; nt < 2; ++nt) {
        float bv = SF[FB_G2B2 + nt * 16 + col];
        v8f c;
#pragma unroll
        for (int r = 0; r < 8; ++r) c[r] = bv;
        c = wmma_bf16(a, frag(SB + OFF_G2W2 + nt * 16 * LDW2, LDW2, 0, lane), c);
        float s = 0.f;
#pragma unroll
        for (int r = 0; r < 8; ++r)
          s += elu_(c[r] + (float)Xt[(r + half * 8) * LDA + nt * 16 + col]);
        if (nt == 0) sum0 += s; else sum1 += s;
      }
    }
  }

  // ---- mean pool: lane-halves -> wave -> workgroup ----
  sum0 += __shfl_xor(sum0, 16, 32);
  sum1 += __shfl_xor(sum1, 16, 32);
  float* red = SF + FB_RED;
  if (lane < 16) {
    red[wave * 32 + lane]      = sum0;
    red[wave * 32 + 16 + lane] = sum1;
  }
  __syncthreads();
  if (tid < 32) {
    float s = 0.f;
#pragma unroll
    for (int w = 0; w < 8; ++w) s += red[w * 32 + tid];
    p.pooled[seg * 32 + tid] = s * (1.0f / 1024.0f);
  }
}

// ---------------------------------------------------------------------------
// Tail: (1024 rows) fm/gm2/out chains + out_lin. ~20 MFLOP total -> scalar
// f32, deliberately NOT inlined/unrolled (runtime-negligible; keeps code small)
// ---------------------------------------------------------------------------
__device__ __attribute__((noinline)) void lin_f(const float* w, const float* b,
                                                const float* x, float* y,
                                                int fi, int fo) {
#pragma clang loop unroll(disable)
  for (int o = 0; o < fo; ++o) {
    float s = b[o];
#pragma clang loop unroll(disable)
    for (int i = 0; i < fi; ++i) s += x[i] * w[i * fo + o];
    y[o] = s;
  }
}
__device__ __attribute__((noinline)) void resblock_f(const float* b1, const float* w1,
                                                     const float* b2, const float* w2,
                                                     int fi, int inter, int fo,
                                                     float* x) {
  float h[32], h2[32];
  lin_f(w1, b1, x, h, fi, inter);
#pragma clang loop unroll(disable)
  for (int i = 0; i < inter; ++i) h[i] = elu_(h[i]);
  lin_f(w2, b2, h, h2, inter, fo);
  if (fi == fo)
#pragma clang loop unroll(disable)
    for (int i = 0; i < fo; ++i) h2[i] += x[i];
#pragma clang loop unroll(disable)
  for (int i = 0; i < fo; ++i) x[i] = elu_(h2[i]);
}

struct TailArgs {
  const float* sf;        // sample_feature (8192 x 64)
  const float* pooled;    // (1024 x 32) from main kernel
  const float* fm[12];    // per block: b1,w1,b2,w2
  const float* gm2[12];
  const float* ob[12];    // 'out' blocks
  const float* olb;       // out_lin.b (2)
  const float* olw;       // out_lin.w (32x2)
  float* out;             // d_out (1024 x 2)
};

__global__ __launch_bounds__(256) void tail_kernel(TailArgs t) {
  int r = blockIdx.x * 256 + threadIdx.x;
  if (r >= 1024) return;

  float xf[64];
#pragma clang loop unroll(disable)
  for (int c = 0; c < 64; ++c) {                 // pf = mean of 8 rows
    float s = 0.f;
    for (int j = 0; j < 8; ++j) s += t.sf[(r * 8 + j) * 64 + c];
    xf[c] = s * 0.125f;
  }
  resblock_f(t.fm[0], t.fm[1], t.fm[2], t.fm[3],   64, 32, 32, xf);
  resblock_f(t.fm[4], t.fm[5], t.fm[6], t.fm[7],   32, 32, 32, xf);
  resblock_f(t.fm[8], t.fm[9], t.fm[10], t.fm[11], 32, 16, 32, xf);

  float xg[64];
#pragma clang loop unroll(disable)
  for (int c = 0; c < 32; ++c) xg[c] = t.pooled[r * 32 + c];
  resblock_f(t.gm2[0], t.gm2[1], t.gm2[2], t.gm2[3],   32, 32, 32, xg);
  resblock_f(t.gm2[4], t.gm2[5], t.gm2[6], t.gm2[7],   32, 32, 32, xg);
  resblock_f(t.gm2[8], t.gm2[9], t.gm2[10], t.gm2[11], 32, 16, 32, xg);

  float cat[64];
#pragma clang loop unroll(disable)
  for (int i = 0; i < 32; ++i) { cat[i] = xf[i]; cat[32 + i] = xg[i]; }
  resblock_f(t.ob[0], t.ob[1], t.ob[2], t.ob[3],   64, 32, 32, cat);
  resblock_f(t.ob[4], t.ob[5], t.ob[6], t.ob[7],   32, 32, 32, cat);
  resblock_f(t.ob[8], t.ob[9], t.ob[10], t.ob[11], 32, 16, 32, cat);

#pragma clang loop unroll(disable)
  for (int k = 0; k < 2; ++k) {
    float s = t.olb[k];
    for (int i = 0; i < 32; ++i) s += cat[i] * t.olw[i * 2 + k];
    t.out[r * 2 + k] = s;
  }
}

// ---------------------------------------------------------------------------
// Input index map (setup_inputs dict order, then jax pytree leaves of params
// with alphabetical dict keys: fm, gm1, gm2, nn, out, out_lin; per linear:
// b before w):
//   0 graph_out   1 sample_node_id   2 sample_node_feature
//   3 sample_id   4 sample_feature
//   5..16 fm   17..28 gm1   29..40 gm2   41..52 nn   53..64 out
//   65 out_lin.b   66 out_lin.w
// ---------------------------------------------------------------------------
extern "C" void kernel_launch(void* const* d_in, const int* in_sizes, int n_in,
                              void* d_out, int out_size, void* d_ws, size_t ws_size,
                              hipStream_t stream) {
  (void)in_sizes; (void)n_in; (void)out_size; (void)ws_size;
  const int FM = 5, GM1 = 17, GM2 = 29, NN = 41, OUT = 53;

  MainArgs m;
  m.x   = (const float*)d_in[2];
  m.nid = (const int*)d_in[1];
  m.g   = (const float*)d_in[0];
  for (int blk = 0; blk < 3; ++blk)
    for (int l = 0; l < 2; ++l) {
      m.b[blk * 2 + l]     = (const float*)d_in[NN  + blk * 4 + l * 2];
      m.w[blk * 2 + l]     = (const float*)d_in[NN  + blk * 4 + l * 2 + 1];
      m.b[6 + blk * 2 + l] = (const float*)d_in[GM1 + blk * 4 + l * 2];
      m.w[6 + blk * 2 + l] = (const float*)d_in[GM1 + blk * 4 + l * 2 + 1];
    }
  m.pooled = (float*)d_ws;

  segment_mlp_kernel<<<1024, 256, SMEM_BYTES, stream>>>(m);

  TailArgs t;
  t.sf     = (const float*)d_in[4];
  t.pooled = (const float*)d_ws;
  for (int j = 0; j < 12; ++j) {
    t.fm[j]  = (const float*)d_in[FM + j];
    t.gm2[j] = (const float*)d_in[GM2 + j];
    t.ob[j]  = (const float*)d_in[OUT + j];
  }
  t.olb = (const float*)d_in[65];
  t.olw = (const float*)d_in[66];
  t.out = (float*)d_out;

  tail_kernel<<<4, 256, 0, stream>>>(t);
}
